// MemoryGate_7988639171118
// MI455X (gfx1250) — matrix operand order repaired and checked by reading, not compiled, not run on hardware
//
#include <hip/hip_runtime.h>

typedef __bf16 v16bf __attribute__((ext_vector_type(16)));
typedef __bf16 v8bf  __attribute__((ext_vector_type(8)));
typedef float  v8f   __attribute__((ext_vector_type(8)));

__device__ __forceinline__ v8f wmma_bf16(v16bf a, v16bf b, v8f c) {
  // 8 args: (neg_a, A, neg_b, B, c_mod, C, reuse_a, reuse_b)
  return __builtin_amdgcn_wmma_f32_16x16x32_bf16(false, a, false, b, (short)0, c,
                                                 false, false);
}

// A fragment: 16x32 bf16 from row-major bf16 source (row stride 64).
// Lane L: row mr + (L%16), half h = L/16:
//   elems 0..7  -> K = kb + 8h + e      ; elems 8..15 -> K = kb + 16 + 8h + e
__device__ __forceinline__ v16bf load_a_frag(const __bf16* base, int mr, int kb,
                                             int lane) {
  const int r = lane & 15, h = lane >> 4;
  const __bf16* p = base + (mr + r) * 64 + kb + h * 8;
  v8bf lo = *(const v8bf*)p;
  v8bf hi = *(const v8bf*)(p + 16);
  v16bf a;
#pragma unroll
  for (int e = 0; e < 8; ++e) { a[e] = lo[e]; a[8 + e] = hi[e]; }
  return a;
}

// Same A fragment but from a row-major f32 source (convert at load).
__device__ __forceinline__ v16bf load_a_frag_f32(const float* base, int mr,
                                                 int kb, int lane) {
  const int r = lane & 15, h = lane >> 4;
  const float* p = base + (mr + r) * 64 + kb + h * 8;
  v16bf a;
#pragma unroll
  for (int e = 0; e < 8; ++e) {
    a[e] = (__bf16)p[e];
    a[8 + e] = (__bf16)p[16 + e];
  }
  return a;
}

// B fragment: 32x16 bf16 from K-major ("Bt") array bt[col*64 + k].
// Lane L: col nc + (L%16); elems 0..7 -> K = kb + 8h + e ; elems 8..15 -> K = kb + 16 + 8h + e
__device__ __forceinline__ v16bf load_b_frag(const __bf16* bt, int nc, int kb,
                                             int lane) {
  const int cl = lane & 15, h = lane >> 4;
  const __bf16* p = bt + (nc + cl) * 64 + kb + h * 8;
  v8bf lo = *(const v8bf*)p;
  v8bf hi = *(const v8bf*)(p + 16);
  v16bf b;
#pragma unroll
  for (int e = 0; e < 8; ++e) { b[e] = lo[e]; b[8 + e] = hi[e]; }
  return b;
}

// C fragment: row = mr + 8h + r, col = nc + (lane%16)
__device__ __forceinline__ void store_rm_bf16(__bf16* d, int mr, int nc,
                                              int lane, v8f c) {
  const int cl = lane & 15, h = lane >> 4;
#pragma unroll
  for (int r = 0; r < 8; ++r) d[(mr + 8 * h + r) * 64 + nc + cl] = (__bf16)c[r];
}

// Transposed C store: contiguous 8 x bf16 per lane -> one ds_store_b128.
__device__ __forceinline__ void store_tr_bf16(__bf16* d, int mr, int nc,
                                              int lane, v8f c) {
  const int cl = lane & 15, h = lane >> 4;
  v8bf t;
#pragma unroll
  for (int r = 0; r < 8; ++r) t[r] = (__bf16)c[r];
  *(v8bf*)(d + (nc + cl) * 64 + mr + 8 * h) = t;
}

__device__ __forceinline__ void store_rm_f32(float* d, int mr, int nc, int lane,
                                             v8f c) {
  const int cl = lane & 15, h = lane >> 4;
#pragma unroll
  for (int r = 0; r < 8; ++r) d[(mr + 8 * h + r) * 64 + nc + cl] = c[r];
}

__device__ __forceinline__ void async_b128(unsigned lds_byte_addr,
                                           unsigned gvoff_bytes,
                                           unsigned long long gbase) {
  asm volatile("global_load_async_to_lds_b128 %0, %1, %2 offset:0" ::"v"(
                   lds_byte_addr),
               "v"(gvoff_bytes), "s"(gbase)
               : "memory");
}

// One-time: W (f32, row-major) -> W^T (bf16) into workspace.
// wsT layout: [0]=WqT, [4096]=WkT, [8192]=WvT  (each 64x64 bf16, n-major)
__global__ __launch_bounds__(256) void prep_weights_kernel(
    const float* __restrict__ wq, const float* __restrict__ wk,
    const float* __restrict__ wv, __bf16* __restrict__ wsT) {
  const int i = blockIdx.x * 256 + threadIdx.x;
  if (i < 4096) {
    const int k = i >> 6, n = i & 63;
    wsT[n * 64 + k] = (__bf16)wq[i];
    wsT[4096 + n * 64 + k] = (__bf16)wk[i];
    wsT[8192 + n * 64 + k] = (__bf16)wv[i];
  }
}

template <bool USE_WS>
__global__ __launch_bounds__(128) void memory_gate_kernel(
    const float* __restrict__ x, const float* __restrict__ wq,
    const float* __restrict__ wk, const float* __restrict__ wv,
    const __bf16* __restrict__ wsT, float* __restrict__ out) {
  __shared__ __align__(128) unsigned char smem[65536];
  float*  sXf = (float*)(smem);            // 16KB f32 X, phases 0-1 (DMA dest)
  __bf16* sWq = (__bf16*)(smem + 16384);   // 8KB W^T, phases 0-1
  __bf16* sWk = (__bf16*)(smem + 24576);   // 8KB W^T, phases 0-1
  __bf16* sWv = (__bf16*)(smem + 32768);   // 8KB W^T, phases 0-1
  float*  sE  = (float*)(smem);            // 16KB, reuses sXf (phases 2-3)
  __bf16* sS  = (__bf16*)(smem + 16384);   // 8KB, reuses sWq (phases 3-4)
  __bf16* sQ  = (__bf16*)(smem + 40960);   // 8KB
  __bf16* sK  = (__bf16*)(smem + 49152);   // 8KB (row-major == Bt for Q.K^T)
  __bf16* sVt = (__bf16*)(smem + 57344);   // 8KB (V^T == Bt for S.V)

  const int tid = threadIdx.x;
  const int lane = tid & 31;
  const int wid = tid >> 5;
  const int mr = wid * 16;
  const size_t sbase = (size_t)blockIdx.x * 4096;

  // ---- Phase 0a: async DMA the f32 X tile (16KB) straight into LDS ----
  {
    const unsigned long long gbase = (unsigned long long)(x + sbase);
    const unsigned goff = (unsigned)tid * 16u;  // bytes, b128 per thread
    const unsigned lX = (unsigned)(unsigned long long)(void*)sXf + goff;
#pragma unroll
    for (int it = 0; it < 8; ++it)
      async_b128(lX + (unsigned)it * 2048u, goff + (unsigned)it * 2048u, gbase);
  }

  // ---- Phase 0b: stage bf16 W^T (24KB) ----
  if (USE_WS) {
    // Pre-converted in workspace: pure DMA, no VALU.
    const unsigned long long gbase = (unsigned long long)wsT;
    const unsigned goff = (unsigned)tid * 16u;
    const unsigned lW = (unsigned)(unsigned long long)(void*)sWq + goff;
#pragma unroll
    for (int it = 0; it < 12; ++it)
      async_b128(lW + (unsigned)it * 2048u, goff + (unsigned)it * 2048u, gbase);
  } else {
    // Fallback: convert + transpose in-block (weights are L2-resident).
    const float4* q4 = (const float4*)wq;
    const float4* k4 = (const float4*)wk;
    const float4* v4 = (const float4*)wv;
#pragma unroll 2
    for (int i = tid; i < 1024; i += 128) {
      const int kk = i >> 4;
      const int nn = (i & 15) * 4;
      const float4 a = q4[i];
      const float4 b = k4[i];
      const float4 c = v4[i];
      sWq[(nn + 0) * 64 + kk] = (__bf16)a.x;
      sWq[(nn + 1) * 64 + kk] = (__bf16)a.y;
      sWq[(nn + 2) * 64 + kk] = (__bf16)a.z;
      sWq[(nn + 3) * 64 + kk] = (__bf16)a.w;
      sWk[(nn + 0) * 64 + kk] = (__bf16)b.x;
      sWk[(nn + 1) * 64 + kk] = (__bf16)b.y;
      sWk[(nn + 2) * 64 + kk] = (__bf16)b.z;
      sWk[(nn + 3) * 64 + kk] = (__bf16)b.w;
      sWv[(nn + 0) * 64 + kk] = (__bf16)c.x;
      sWv[(nn + 1) * 64 + kk] = (__bf16)c.y;
      sWv[(nn + 2) * 64 + kk] = (__bf16)c.z;
      sWv[(nn + 3) * 64 + kk] = (__bf16)c.w;
    }
  }
  asm volatile("s_wait_asynccnt 0" ::: "memory");  // my DMA landed
  __syncthreads();                                 // everyone's landed

  // ---- Phase 1: Q = X Wq, K = X Wk, V = X Wv (per-wave 16x64 strip) ----
  {
    v16bf a0 = load_a_frag_f32(sXf, mr, 0, lane);
    v16bf a1 = load_a_frag_f32(sXf, mr, 32, lane);
#pragma unroll
    for (int j = 0; j < 4; ++j) {
      v8f c = {};
      c = wmma_bf16(a0, load_b_frag(sWq, j * 16, 0, lane), c);
      c = wmma_bf16(a1, load_b_frag(sWq, j * 16, 32, lane), c);
      store_rm_bf16(sQ, mr, j * 16, lane, c);
    }
#pragma unroll
    for (int j = 0; j < 4; ++j) {
      v8f c = {};
      c = wmma_bf16(a0, load_b_frag(sWk, j * 16, 0, lane), c);
      c = wmma_bf16(a1, load_b_frag(sWk, j * 16, 32, lane), c);
      store_rm_bf16(sK, mr, j * 16, lane, c);
    }
#pragma unroll
    for (int j = 0; j < 4; ++j) {
      v8f c = {};
      c = wmma_bf16(a0, load_b_frag(sWv, j * 16, 0, lane), c);
      c = wmma_bf16(a1, load_b_frag(sWv, j * 16, 32, lane), c);
      store_tr_bf16(sVt, mr, j * 16, lane, c);
    }
  }
  __syncthreads();

  // ---- Phase 2: E = Q K^T (f32 in LDS; sK row-major is the Bt layout) ----
  {
    v16bf a0 = load_a_frag(sQ, mr, 0, lane);
    v16bf a1 = load_a_frag(sQ, mr, 32, lane);
#pragma unroll
    for (int j = 0; j < 4; ++j) {
      v8f c = {};
      c = wmma_bf16(a0, load_b_frag(sK, j * 16, 0, lane), c);
      c = wmma_bf16(a1, load_b_frag(sK, j * 16, 32, lane), c);
      store_rm_f32(sE, mr, j * 16, lane, c);
    }
  }
  __syncthreads();

  // ---- Phase 3: per-row top-3 threshold; score = (E>=kth) ? relu(E) : 0 ----
  if (tid < 64) {
    const float* e = sE + tid * 64;
    const float4* e4 = (const float4*)e;
    float m1 = -3.4e38f, m2 = m1, m3 = m1;
#pragma unroll 2
    for (int s4 = 0; s4 < 16; ++s4) {
      const float4 v4 = e4[s4];
      const float vs[4] = {v4.x, v4.y, v4.z, v4.w};
#pragma unroll
      for (int c = 0; c < 4; ++c) {
        const float v = vs[c];
        if (v > m1) { m3 = m2; m2 = m1; m1 = v; }
        else if (v > m2) { m3 = m2; m2 = v; }
        else if (v > m3) { m3 = v; }
      }
    }
    __bf16* srow = sS + tid * 64;
#pragma unroll 2
    for (int s8 = 0; s8 < 8; ++s8) {
      v8bf t;
#pragma unroll
      for (int c = 0; c < 8; ++c) {
        const float v = e[s8 * 8 + c];
        t[c] = (__bf16)((v >= m3) ? fmaxf(v, 0.0f) : 0.0f);
      }
      *(v8bf*)(srow + s8 * 8) = t;
    }
  }
  __syncthreads();

  // ---- Phase 4: O = S V -> global (f32) ----
  {
    v16bf a0 = load_a_frag(sS, mr, 0, lane);
    v16bf a1 = load_a_frag(sS, mr, 32, lane);
    float* op = out + sbase;
    const int cl = lane & 15, h = lane >> 4;
#pragma unroll
    for (int j = 0; j < 4; ++j) {
      v8f c = {};
      c = wmma_bf16(a0, load_b_frag(sVt, j * 16, 0, lane), c);
      c = wmma_bf16(a1, load_b_frag(sVt, j * 16, 32, lane), c);
#pragma unroll
      for (int r = 0; r < 8; ++r)
        op[(mr + 8 * h + r) * 64 + j * 16 + cl] = c[r];
    }
  }
}

extern "C" void kernel_launch(void* const* d_in, const int* in_sizes, int n_in,
                              void* d_out, int out_size, void* d_ws,
                              size_t ws_size, hipStream_t stream) {
  const float* x = (const float*)d_in[0];
  const float* wq = (const float*)d_in[1];
  const float* wk = (const float*)d_in[2];
  const float* wv = (const float*)d_in[3];
  float* out = (float*)d_out;
  const int nslices = in_sizes[0] / 4096;  // B*N = 8192 (T=H=64 per slice)
  if (nslices <= 0) return;

  if (ws_size >= 24576 && d_ws != nullptr) {
    __bf16* wsT = (__bf16*)d_ws;
    hipLaunchKernelGGL(prep_weights_kernel, dim3(16), dim3(256), 0, stream, wq,
                       wk, wv, wsT);
    hipLaunchKernelGGL(memory_gate_kernel<true>, dim3(nslices), dim3(128), 0,
                       stream, x, wq, wk, wv, (const __bf16*)wsT, out);
  } else {
    hipLaunchKernelGGL(memory_gate_kernel<false>, dim3(nslices), dim3(128), 0,
                       stream, x, wq, wk, wv, (const __bf16*)nullptr, out);
  }
}
